// SupplyChainDetector_16947940950503
// MI455X (gfx1250) — matrix operand order repaired and checked by reading, not compiled, not run on hardware
//
#include <hip/hip_runtime.h>
#include <hip/hip_bf16.h>

typedef _Float16 half_t;
typedef __attribute__((ext_vector_type(16))) _Float16 v16h;
typedef __attribute__((ext_vector_type(8)))  _Float16 v8h;
typedef __attribute__((ext_vector_type(8)))  float    v8f;
typedef __attribute__((ext_vector_type(4)))  int      v4i;

enum { BB = 256, WW = 256, NN = 128, KK = 7, HH = 64, G3 = 192, WPAD = WW + 6 };
#define ALPHA 0.2f

#if defined(__has_builtin)
#if __has_builtin(__builtin_amdgcn_global_load_async_to_lds_b128)
#define GRU_ASYNC_LDS 1
#endif
#endif

#if defined(GRU_ASYNC_LDS)
typedef __attribute__((address_space(1))) v4i gv4i_t;  // global int4
typedef __attribute__((address_space(3))) v4i lv4i_t;  // LDS int4
#endif

// ---------------- WMMA fragment helpers (layouts per CDNA5 ISA 7.12.2) -----

// A: 16x32 f16, row-major tile, leading dim ld (halves). Requires 16B align.
__device__ inline v16h frag_a_load(const half_t* __restrict__ base, int ld) {
  int lane = threadIdx.x & 31;
  int row  = lane & 15;
  int koff = (lane >> 4) << 3;  // 0 or 8
  const half_t* p = base + (size_t)row * ld + koff;
  v8h lo = *(const v8h*)(p);
  v8h hi = *(const v8h*)(p + 16);
  return __builtin_shufflevector(lo, hi, 0,1,2,3,4,5,6,7,8,9,10,11,12,13,14,15);
}

// B: 32x16 f16. lane L holds column N=L%16; K = 8h+s (s<8), 16+8h+s-8 (s>=8).
__device__ inline v16h frag_b_load(const half_t* __restrict__ base, int ld) {
  int lane = threadIdx.x & 31;
  int col  = lane & 15;
  int koff = (lane >> 4) << 3;
  v16h r;
#pragma unroll
  for (int s = 0; s < 8; ++s) {
    r[s]     = base[(size_t)(koff + s) * ld + col];
    r[8 + s] = base[(size_t)(16 + koff + s) * ld + col];
  }
  return r;
}

__device__ inline v8f wmma_f16(v16h a, v16h b, v8f c) {
  return __builtin_amdgcn_wmma_f32_16x16x32_f16(false, a, false, b, (short)0, c,
                                                false, false);
}

__device__ inline float sigm(float x) { return 1.f / (1.f + __expf(-x)); }

// ---------------- prep kernels ---------------------------------------------

// xpT[b][wp][ci] = x[b][wp-3][ci], zero padded 3 rows each side (f16)
__global__ void k_pack_x(const float* __restrict__ x, half_t* __restrict__ xpT) {
  size_t idx = (size_t)blockIdx.x * blockDim.x + threadIdx.x;
  size_t total = (size_t)BB * WPAD * NN;
  if (idx >= total) return;
  int ci = (int)(idx % NN);
  size_t r = idx / NN;
  int wp = (int)(r % WPAD);
  int b  = (int)(r / WPAD);
  float v = (wp >= 3 && wp < WW + 3)
                ? x[((size_t)b * WW + (wp - 3)) * NN + ci]
                : 0.f;
  xpT[idx] = (half_t)v;
}

// cwT[k][ci][co] = conv_w[co][ci][k]  (f16)
__global__ void k_pack_convw(const float* __restrict__ cw, half_t* __restrict__ cwT) {
  int idx = blockIdx.x * blockDim.x + threadIdx.x;
  if (idx >= KK * NN * NN) return;
  int co = idx % NN;
  int t  = idx / NN;
  int ci = t % NN;
  int k  = t / NN;
  cwT[idx] = (half_t)cw[((size_t)co * NN + ci) * KK + k];
}

// dst[n][g] = src[g][n]  (192 g-columns, n_rows = cols of src)
__global__ void k_pack_T(const float* __restrict__ src, half_t* __restrict__ dst,
                         int n_rows) {
  int idx = blockIdx.x * blockDim.x + threadIdx.x;
  if (idx >= n_rows * G3) return;
  int g = idx % G3;
  int n = idx / G3;
  dst[idx] = (half_t)src[(size_t)g * n_rows + n];
}

// fa[win] = sum_wout fc_w[wout][win]*a2[wout] ; fa[WW] = fc_b . a2
__global__ void k_fa(const float* __restrict__ fc_w, const float* __restrict__ fc_b,
                     const float* __restrict__ attn_a, float* __restrict__ fa) {
  int win = threadIdx.x;
  float acc = 0.f;
  for (int wo = 0; wo < WW; ++wo) acc += fc_w[(size_t)wo * WW + win] * attn_a[WW + wo];
  fa[win] = acc;
  if (win == 0) {
    float c = 0.f;
    for (int wo = 0; wo < WW; ++wo) c += fc_b[wo] * attn_a[WW + wo];
    fa[WW] = c;
  }
}

// ---------------- conv1d (implicit GEMM, WMMA, fully aligned) ---------------
// grid (B, 4, 2), block 128 (4 waves). wave: 16 w x 64 co tile.
// M = w (time), N = co, K = (ci, tap). Tap shift moves whole A rows.
__global__ void k_conv(const half_t* __restrict__ xpT, const half_t* __restrict__ cwT,
                       const float* __restrict__ conv_b, half_t* __restrict__ xn) {
  int b = blockIdx.x, wg = blockIdx.y, cg = blockIdx.z;
  int wave = threadIdx.x >> 5, lane = threadIdx.x & 31;
  int w0  = wg * 64 + wave * 16;
  int cobase = cg * 64;
  const half_t* xb = xpT + (size_t)b * WPAD * NN;
  v8f acc[4] = {};
#pragma unroll
  for (int k = 0; k < KK; ++k) {
    for (int ci0 = 0; ci0 < NN; ci0 += 32) {
      v16h a = frag_a_load(xb + (size_t)(w0 + k) * NN + ci0, NN);
      const half_t* bb = cwT + ((size_t)k * NN + ci0) * NN + cobase;
#pragma unroll
      for (int nt = 0; nt < 4; ++nt) {
        v16h bf = frag_b_load(bb + nt * 16, NN);
        acc[nt] = wmma_f16(a, bf, acc[nt]);
      }
    }
  }
  int n = lane & 15, mb = (lane >> 4) * 8;
#pragma unroll
  for (int nt = 0; nt < 4; ++nt) {
    int co = cobase + nt * 16 + n;
    float bias = conv_b[co];
    v8h o = {};
#pragma unroll
    for (int q = 0; q < 8; ++q) {
      float v = acc[nt][q] + bias;
      o[q] = (half_t)(v > 0.f ? v : 0.f);
    }
    *(v8h*)(xn + ((size_t)b * NN + co) * WW + w0 + mb) = o;
  }
}

// ---------------- s1 / s2 (dot products; fc GEMM folded into fa) ------------
__global__ void k_s12(const half_t* __restrict__ xn, const float* __restrict__ attn_a,
                      const float* __restrict__ fa, float* __restrict__ s1,
                      float* __restrict__ s2) {
  int idx = blockIdx.x * blockDim.x + threadIdx.x;  // b*NN + i
  if (idx >= BB * NN) return;
  const half_t* row = xn + (size_t)idx * WW;
  float a1 = 0.f, a2 = 0.f;
  for (int w = 0; w < WW; ++w) {
    float xv = (float)row[w];
    a1 += xv * attn_a[w];
    a2 += xv * fa[w];
  }
  s1[idx] = a1;
  s2[idx] = a2 + fa[WW];
}

// ---------------- leaky-relu + row softmax → att (f16) ----------------------
__global__ void k_attn(const float* __restrict__ s1, const float* __restrict__ s2,
                       half_t* __restrict__ att) {
  int row  = blockIdx.x * (blockDim.x >> 5) + (threadIdx.x >> 5);  // b*NN+i
  int lane = threadIdx.x & 31;
  if (row >= BB * NN) return;
  int b = row / NN;
  float s1v = s1[row];
  const float* s2b = s2 + (size_t)b * NN;
  float e[4];
  float mx = -1e30f;
#pragma unroll
  for (int t = 0; t < 4; ++t) {
    float v = s1v + s2b[lane + 32 * t];
    v = v > 0.f ? v : ALPHA * v;
    e[t] = v;
    mx = fmaxf(mx, v);
  }
  for (int off = 16; off; off >>= 1) mx = fmaxf(mx, __shfl_xor(mx, off, 32));
  float sum = 0.f;
#pragma unroll
  for (int t = 0; t < 4; ++t) { e[t] = __expf(e[t] - mx); sum += e[t]; }
  for (int off = 16; off; off >>= 1) sum += __shfl_xor(sum, off, 32);
  float inv = 1.f / sum;
#pragma unroll
  for (int t = 0; t < 4; ++t)
    att[(size_t)row * NN + lane + 32 * t] = (half_t)(e[t] * inv);
}

// ---------------- hout = sigmoid(att @ xn), stored transposed (B,W,N) -------
// grid (B, 2, 4), block 128. wave: 16 i x 64 w tile.
__global__ void k_apply(const half_t* __restrict__ att, const half_t* __restrict__ xn,
                        half_t* __restrict__ houtT) {
  int b = blockIdx.x, ig = blockIdx.y, wg = blockIdx.z;
  int wave = threadIdx.x >> 5, lane = threadIdx.x & 31;
  int i0 = ig * 64 + wave * 16;
  int w0 = wg * 64;
  const half_t* attb = att + (size_t)b * NN * NN;
  const half_t* xnb  = xn + (size_t)b * NN * WW;
  v8f acc[4] = {};
  for (int j0 = 0; j0 < NN; j0 += 32) {
    v16h a = frag_a_load(attb + (size_t)i0 * NN + j0, NN);
#pragma unroll
    for (int nt = 0; nt < 4; ++nt) {
      v16h bf = frag_b_load(xnb + (size_t)j0 * WW + (w0 + nt * 16), WW);
      acc[nt] = wmma_f16(a, bf, acc[nt]);
    }
  }
  int n = lane & 15, mb = (lane >> 4) * 8;
#pragma unroll
  for (int nt = 0; nt < 4; ++nt) {
    v8h o = {};
#pragma unroll
    for (int q = 0; q < 8; ++q) o[q] = (half_t)sigm(acc[nt][q]);
    *(v8h*)(houtT + ((size_t)b * WW + (w0 + nt * 16 + n)) * NN + i0 + mb) = o;
  }
}

// ---------------- gi0 = houtT @ Wih0^T + bih0, fragment-order output --------
// M = batch (16 rows per btile), N = 192 gates, one wave per timestep t.
// Output gi0f[((t*16+btile)*12+ct)*256 + lane*8 + q] — exactly the per-lane
// D-fragment the GRU wave consumes, so the scan reads 12 x v8h per step.
// grid (16, 64), block 128 (4 waves).
__global__ void k_gi0(const half_t* __restrict__ houtT, const half_t* __restrict__ wih0T,
                      const float* __restrict__ bih0, half_t* __restrict__ gi0f) {
  int btile = blockIdx.x;
  int wave = threadIdx.x >> 5, lane = threadIdx.x & 31;
  int t = blockIdx.y * 4 + wave;
  int b0 = btile * 16;
  const half_t* A = houtT + ((size_t)b0 * WW + t) * NN;  // row stride WW*NN
  v8f acc[12] = {};
  for (int n0 = 0; n0 < NN; n0 += 32) {
    v16h a = frag_a_load(A + n0, WW * NN);
#pragma unroll
    for (int ct = 0; ct < 12; ++ct) {
      v16h bf = frag_b_load(wih0T + (size_t)n0 * G3 + ct * 16, G3);
      acc[ct] = wmma_f16(a, bf, acc[ct]);
    }
  }
  int n = lane & 15;
#pragma unroll
  for (int ct = 0; ct < 12; ++ct) {
    float bias = bih0[ct * 16 + n];
    v8h o = {};
#pragma unroll
    for (int q = 0; q < 8; ++q) o[q] = (half_t)(acc[ct][q] + bias);
    *(v8h*)(gi0f + (((size_t)t * 16 + btile) * 12 + ct) * 256 + lane * 8) = o;
  }
}

// ---------------- persistent 2-layer GRU scan + head ------------------------
// 16 blocks x 1 wave; each wave owns 16 batch rows; weights resident in LDS.
__global__ void __launch_bounds__(32) k_gru(
    const half_t* __restrict__ gi0f, const half_t* __restrict__ whh0T,
    const half_t* __restrict__ wih1T, const half_t* __restrict__ whh1T,
    const float* __restrict__ bhh0, const float* __restrict__ bih1,
    const float* __restrict__ bhh1, const float* __restrict__ head_w1,
    const float* __restrict__ head_b1, const float* __restrict__ head_w2,
    const float* __restrict__ head_b2, float* __restrict__ out) {
  __shared__ half_t sW[3 * HH * G3];   // Whh0T | Wih1T | Whh1T  (64x192 each)
  __shared__ half_t sh0[16 * HH];
  __shared__ half_t sh1[16 * HH];
  int lane = threadIdx.x;
  int btile = blockIdx.x;
  int b0 = btile * 16;

#if defined(GRU_ASYNC_LDS)
  {
    const half_t* gsrc[3] = {whh0T, wih1T, whh1T};
#pragma unroll
    for (int m = 0; m < 3; ++m)
      for (int i = lane * 8; i < HH * G3; i += 32 * 8)
        __builtin_amdgcn_global_load_async_to_lds_b128(
            (gv4i_t*)(uintptr_t)(gsrc[m] + i),
            (lv4i_t*)(uint32_t)(uintptr_t)(sW + m * HH * G3 + i), 0, 0);
    asm volatile("s_wait_asynccnt 0" ::: "memory");
  }
#else
  for (int i = lane; i < HH * G3; i += 32) {
    sW[i]               = whh0T[i];
    sW[HH * G3 + i]     = wih1T[i];
    sW[2 * HH * G3 + i] = whh1T[i];
  }
#endif
  for (int i = lane; i < 16 * HH; i += 32) { sh0[i] = (half_t)0.f; sh1[i] = (half_t)0.f; }
  __syncthreads();

  const half_t* sWhh0 = sW;
  const half_t* sWih1 = sW + HH * G3;
  const half_t* sWhh1 = sW + 2 * HH * G3;

  int n = lane & 15, mb = (lane >> 4) * 8;
  float vbhh0[12], vbih1[12], vbhh1[12];
#pragma unroll
  for (int ct = 0; ct < 12; ++ct) {
    int g = ct * 16 + n;
    vbhh0[ct] = bhh0[g]; vbih1[ct] = bih1[g]; vbhh1[ct] = bhh1[g];
  }

  // preload step-0 input fragments
  v8h giv[12];
  {
    const half_t* gp = gi0f + ((size_t)btile * 12) * 256 + lane * 8;
#pragma unroll
    for (int ct = 0; ct < 12; ++ct) giv[ct] = *(const v8h*)(gp + ct * 256);
  }

  for (int t = 0; t < WW; ++t) {
    // ---- layer 0: gh0 = h0 @ Whh0^T + bhh0
    v16h a0 = frag_a_load(sh0, HH);
    v16h a1 = frag_a_load(sh0 + 32, HH);
    v8f gh[12];
#pragma unroll
    for (int ct = 0; ct < 12; ++ct) {
      v8f c = {};
      c = wmma_f16(a0, frag_b_load(sWhh0 + ct * 16, G3), c);
      c = wmma_f16(a1, frag_b_load(sWhh0 + (size_t)32 * G3 + ct * 16, G3), c);
      gh[ct] = c;
    }
#pragma unroll
    for (int ht = 0; ht < 4; ++ht) {
#pragma unroll
      for (int q = 0; q < 8; ++q) {
        int brow = mb + q;
        float rg = sigm((float)giv[ht][q] + gh[ht][q] + vbhh0[ht]);
        float zg = sigm((float)giv[4 + ht][q] + gh[4 + ht][q] + vbhh0[4 + ht]);
        float ng = tanhf((float)giv[8 + ht][q] + rg * (gh[8 + ht][q] + vbhh0[8 + ht]));
        float hp = (float)sh0[brow * HH + ht * 16 + n];
        sh0[brow * HH + ht * 16 + n] = (half_t)((1.f - zg) * ng + zg * hp);
      }
    }
    asm volatile("" ::: "memory");  // same-wave DS ops are HW-ordered; block
                                    // only compiler reordering (no s_barrier)
    // ---- prefetch next step's input fragments (overlaps layer-1 WMMAs)
    v8h ginext[12] = {};
    if (t + 1 < WW) {
      const half_t* gp = gi0f + (((size_t)(t + 1) * 16 + btile) * 12) * 256 + lane * 8;
#pragma unroll
      for (int ct = 0; ct < 12; ++ct) ginext[ct] = *(const v8h*)(gp + ct * 256);
    }
    // ---- layer 1: gi1 = h0' @ Wih1^T + bih1 ; gh1 = h1 @ Whh1^T + bhh1
    v16h b0f = frag_a_load(sh0, HH);
    v16h b1f = frag_a_load(sh0 + 32, HH);
    v16h c0f = frag_a_load(sh1, HH);
    v16h c1f = frag_a_load(sh1 + 32, HH);
    v8f gi1[12], gh1[12];
#pragma unroll
    for (int ct = 0; ct < 12; ++ct) {
      v8f c = {};
      c = wmma_f16(b0f, frag_b_load(sWih1 + ct * 16, G3), c);
      c = wmma_f16(b1f, frag_b_load(sWih1 + (size_t)32 * G3 + ct * 16, G3), c);
      gi1[ct] = c;
      v8f d = {};
      d = wmma_f16(c0f, frag_b_load(sWhh1 + ct * 16, G3), d);
      d = wmma_f16(c1f, frag_b_load(sWhh1 + (size_t)32 * G3 + ct * 16, G3), d);
      gh1[ct] = d;
    }
#pragma unroll
    for (int ht = 0; ht < 4; ++ht) {
#pragma unroll
      for (int q = 0; q < 8; ++q) {
        int brow = mb + q;
        float rg = sigm(gi1[ht][q] + vbih1[ht] + gh1[ht][q] + vbhh1[ht]);
        float zg = sigm(gi1[4 + ht][q] + vbih1[4 + ht] + gh1[4 + ht][q] + vbhh1[4 + ht]);
        float ng = tanhf(gi1[8 + ht][q] + vbih1[8 + ht] +
                         rg * (gh1[8 + ht][q] + vbhh1[8 + ht]));
        float hp = (float)sh1[brow * HH + ht * 16 + n];
        sh1[brow * HH + ht * 16 + n] = (half_t)((1.f - zg) * ng + zg * hp);
      }
    }
    asm volatile("" ::: "memory");
#pragma unroll
    for (int ct = 0; ct < 12; ++ct) giv[ct] = ginext[ct];
  }
  // ---- head: out = sigmoid(relu(h1 @ W1^T + b1) @ W2^T + b2)
  if (lane < 16) {
    float z[32];
#pragma unroll
    for (int j = 0; j < 32; ++j) {
      float acc = head_b1[j];
      for (int h = 0; h < HH; ++h)
        acc += (float)sh1[lane * HH + h] * head_w1[(size_t)j * HH + h];
      z[j] = acc > 0.f ? acc : 0.f;
    }
    float o = head_b2[0];
#pragma unroll
    for (int j = 0; j < 32; ++j) o += z[j] * head_w2[j];
    out[b0 + lane] = sigm(o);
  }
}

// ---------------- host launch ----------------------------------------------
extern "C" void kernel_launch(void* const* d_in, const int* in_sizes, int n_in,
                              void* d_out, int out_size, void* d_ws, size_t ws_size,
                              hipStream_t stream) {
  (void)in_sizes; (void)n_in; (void)out_size; (void)ws_size;
  const float* x       = (const float*)d_in[0];
  const float* conv_w  = (const float*)d_in[1];
  const float* conv_b  = (const float*)d_in[2];
  const float* fc_w    = (const float*)d_in[3];
  const float* fc_b    = (const float*)d_in[4];
  const float* attn_a  = (const float*)d_in[5];
  const float* Wih0    = (const float*)d_in[6];
  const float* Whh0    = (const float*)d_in[7];
  const float* bih0    = (const float*)d_in[8];
  const float* bhh0    = (const float*)d_in[9];
  const float* Wih1    = (const float*)d_in[10];
  const float* Whh1    = (const float*)d_in[11];
  const float* bih1    = (const float*)d_in[12];
  const float* bhh1    = (const float*)d_in[13];
  const float* head_w1 = (const float*)d_in[14];
  const float* head_b1 = (const float*)d_in[15];
  const float* head_w2 = (const float*)d_in[16];
  const float* head_b2 = (const float*)d_in[17];
  float* out = (float*)d_out;

  uint8_t* ws = (uint8_t*)d_ws;
  size_t off = 0;
  auto take = [&](size_t bytes) -> void* {
    void* p = ws + off;
    off += (bytes + 255) & ~(size_t)255;
    return p;
  };
  half_t* xpT   = (half_t*)take((size_t)BB * WPAD * NN * 2);
  half_t* cwT   = (half_t*)take((size_t)KK * NN * NN * 2);
  float*  fa    = (float*)take((WW + 1) * 4);
  half_t* wih0T = (half_t*)take((size_t)NN * G3 * 2);
  half_t* whh0T = (half_t*)take((size_t)HH * G3 * 2);
  half_t* wih1T = (half_t*)take((size_t)HH * G3 * 2);
  half_t* whh1T = (half_t*)take((size_t)HH * G3 * 2);
  half_t* xn    = (half_t*)take((size_t)BB * NN * WW * 2);
  float*  s1    = (float*)take((size_t)BB * NN * 4);
  float*  s2    = (float*)take((size_t)BB * NN * 4);
  half_t* att   = (half_t*)take((size_t)BB * NN * NN * 2);
  half_t* houtT = (half_t*)take((size_t)BB * WW * NN * 2);
  half_t* gi0f  = (half_t*)take((size_t)WW * 16 * 12 * 256 * 2);

  {
    size_t total = (size_t)BB * WPAD * NN;
    k_pack_x<<<(unsigned)((total + 255) / 256), 256, 0, stream>>>(x, xpT);
  }
  k_pack_convw<<<(KK * NN * NN + 255) / 256, 256, 0, stream>>>(conv_w, cwT);
  k_pack_T<<<(NN * G3 + 255) / 256, 256, 0, stream>>>(Wih0, wih0T, NN);
  k_pack_T<<<(HH * G3 + 255) / 256, 256, 0, stream>>>(Whh0, whh0T, HH);
  k_pack_T<<<(HH * G3 + 255) / 256, 256, 0, stream>>>(Wih1, wih1T, HH);
  k_pack_T<<<(HH * G3 + 255) / 256, 256, 0, stream>>>(Whh1, whh1T, HH);
  k_fa<<<1, 256, 0, stream>>>(fc_w, fc_b, attn_a, fa);

  k_conv<<<dim3(BB, 4, 2), 128, 0, stream>>>(xpT, cwT, conv_b, xn);
  k_s12<<<(BB * NN + 255) / 256, 256, 0, stream>>>(xn, attn_a, fa, s1, s2);
  k_attn<<<(BB * NN) / 4, 128, 0, stream>>>(s1, s2, att);
  k_apply<<<dim3(BB, 2, 4), 128, 0, stream>>>(att, xn, houtT);
  k_gi0<<<dim3(16, 64), 128, 0, stream>>>(houtT, wih0T, bih0, gi0f);
  k_gru<<<BB / 16, 32, 0, stream>>>(gi0f, whh0T, wih1T, whh1T, bhh0, bih1, bhh1,
                                    head_w1, head_b1, head_w2, head_b2, out);
}
